// DGI_64750926954672
// MI455X (gfx1250) — matrix operand order repaired and checked by reading, not compiled, not run on hardware
//
#include <hip/hip_runtime.h>

typedef __attribute__((ext_vector_type(16))) __bf16 v16bf;
typedef __attribute__((ext_vector_type(8)))  __bf16 v8bf;
typedef __attribute__((ext_vector_type(8)))  float  v8f;
typedef __attribute__((ext_vector_type(4)))  float  v4f;

// ---------------------------------------------------------------------------
// Degree / normalization
// ---------------------------------------------------------------------------
__global__ void deg_kernel(const int* __restrict__ dst, float* __restrict__ deg, int E) {
    int e = blockIdx.x * blockDim.x + threadIdx.x;
    if (e < E) atomicAdd(&deg[dst[e]], 1.0f);
}

__global__ void dinv_kernel(float* __restrict__ deg, int n) {
    int i = blockIdx.x * blockDim.x + threadIdx.x;
    if (i < n) {
        float d = deg[i] + 1.0f;   // + self loop
        deg[i] = rsqrtf(d);        // deg >= 1 always
    }
}

// wrow[d] = sum_e Wd[d,e]
__global__ void wrow_kernel(const float* __restrict__ Wd, float* __restrict__ wrow) {
    int d = threadIdx.x;  // 64 threads
    float s = 0.f;
    #pragma unroll 8
    for (int e = 0; e < 64; ++e) s += Wd[d * 64 + e];
    wrow[d] = s;
}

// ---------------------------------------------------------------------------
// WMMA GEMM: C[nrows x 64] = A[nrows x K] @ W[K x 64], bf16 inputs, f32 acc.
// Block = 128 threads = 4 waves; block computes a 16-row x 64-col output slab,
// wave w owns column tile w. Both operands staged through LDS so each lane's
// WMMA fragment is a contiguous run -> ds_load_b128 (no scalar u16 gathers):
//   Bs[n*K + k] = bf16(W[k*64 + n])   (transposed weights, once per block)
//   As[m*K + k] = bf16(A[(row0+m)*K + k])  (A tile shared by all 4 waves)
// Fragment layouts per CDNA5 ISA 7.12.2 (wave32):
//   A 16x32 bf16: lane L -> row M = L&15, kbase = (L<16)?0:8,
//                 elems 0..7 = K kbase..kbase+7, elems 8..15 = K kbase+16..+23
//   B 32x16 bf16: lane L -> col N = L&15, kb = (L<16)?0:8,
//                 elems 0..7 = K kb..kb+7, elems 8..15 = K kb+16..+23
//   C 16x16 f32 : lane L -> col N = L&15, VGPR r -> row M = r + ((L<16)?0:8)
// Epilogue: single uniform full-tile branch, base pointer + immediate offsets
// so the hot path is 8 straight coalesced global_store_b32.
// ---------------------------------------------------------------------------
template <int K>
__global__ __launch_bounds__(128)
void gemm_bf16_wmma(const float* __restrict__ A, const float* __restrict__ W,
                    float* __restrict__ C, int nrows) {
    __shared__ __align__(32) __bf16 Bs[64 * K];  // W transposed, bf16
    __shared__ __align__(32) __bf16 As[16 * K];  // A tile, bf16
    const int t    = threadIdx.x;
    const int row0 = blockIdx.x * 16;

    // stage W^T (coalesced global reads; tiny, once per block)
    for (int i = t; i < K * 64; i += 128) {
        int k = i >> 6, n = i & 63;
        Bs[n * K + k] = (__bf16)W[i];
    }
    // stage A tile (coalesced; rows clamped so EXEC stays full later)
    for (int i = t; i < 16 * K; i += 128) {
        int m = i / K;
        int row = row0 + m;
        if (row >= nrows) row = nrows - 1;
        As[i] = (__bf16)A[(size_t)row * K + (i % K)];
    }
    __syncthreads();

    const int wave  = t >> 5;          // output column tile 0..3
    const int lane  = t & 31;
    const int m     = lane & 15;
    const int abase = (lane < 16) ? 0 : 8;   // A K sub-base
    const int kb    = (lane < 16) ? 0 : 8;   // B K sub-base
    const int col   = (wave << 4) + m;       // B/C column

    v8f acc = {};
    #pragma unroll
    for (int kk = 0; kk < K; kk += 32) {
        v8bf alo = *(const v8bf*)(&As[m * K + kk + abase]);        // 16B aligned
        v8bf ahi = *(const v8bf*)(&As[m * K + kk + abase + 16]);
        v8bf blo = *(const v8bf*)(&Bs[col * K + kk + kb]);         // 16B aligned
        v8bf bhi = *(const v8bf*)(&Bs[col * K + kk + kb + 16]);
        v16bf a, b;
        #pragma unroll
        for (int j = 0; j < 8; ++j) {
            a[j] = alo[j]; a[8 + j] = ahi[j];
            b[j] = blo[j]; b[8 + j] = bhi[j];
        }
        acc = __builtin_amdgcn_wmma_f32_16x16x32_bf16(
            /*neg_a=*/false, a, /*neg_b=*/false, b,
            /*c_mod=*/(short)0, acc, /*reuse_a=*/false, /*reuse_b=*/false);
    }

    const int mrow0 = (lane < 16) ? 0 : 8;
    float* cp = C + (size_t)(row0 + mrow0) * 64 + col;
    if (row0 + 16 <= nrows) {
        // full tile (always taken when nrows % 16 == 0): straight-line stores
        #pragma unroll
        for (int r = 0; r < 8; ++r) cp[r * 64] = acc[r];
    } else {
        #pragma unroll
        for (int r = 0; r < 8; ++r)
            if (row0 + mrow0 + r < nrows) cp[r * 64] = acc[r];
    }
}

// ---------------------------------------------------------------------------
// Edge scatter: out[dst] += h[src] * dinv[src]*dinv[dst]
// 16 threads per edge, float4 payload each (64 features).
// ---------------------------------------------------------------------------
__global__ void scatter_kernel(const float* __restrict__ h, const int* __restrict__ src,
                               const int* __restrict__ dst, const float* __restrict__ dinv,
                               float* __restrict__ out, int E) {
    long long tid = (long long)blockIdx.x * blockDim.x + threadIdx.x;
    int e = (int)(tid >> 4);
    if (e >= E) return;
    int f = ((int)tid & 15) * 4;
    int s = src[e], d = dst[e];
    float nrm = dinv[s] * dinv[d];
    v4f v = *(const v4f*)(h + (size_t)s * 64 + f);
    float* o = out + (size_t)d * 64 + f;
    atomicAdd(o + 0, v.x * nrm);
    atomicAdd(o + 1, v.y * nrm);
    atomicAdd(o + 2, v.z * nrm);
    atomicAdd(o + 3, v.w * nrm);
}

// agg += self-loop contribution + bias, optional ReLU (in place)
__global__ void postproc_kernel(float* __restrict__ agg, const float* __restrict__ hm,
                                const float* __restrict__ dinv, const float* __restrict__ bias,
                                long long total, int relu) {
    long long tid = (long long)blockIdx.x * blockDim.x + threadIdx.x;
    if (tid >= total) return;
    int i = (int)(tid >> 6), f = (int)tid & 63;
    float di = dinv[i];
    float v = agg[tid] + hm[tid] * di * di + bias[f];
    if (relu) v = fmaxf(v, 0.f);
    agg[tid] = v;
}

// ---------------------------------------------------------------------------
// Readout: c[i] = sigmoid(mean_f h1[i,f]); sc[i] = c[i]*dot(h[i], wrow) + bd
// phase 0: compute c from h, store c, write out[i]
// phase 1: reuse c, write out[n+i]
// ---------------------------------------------------------------------------
__global__ void score_kernel(const float* __restrict__ h, const float* __restrict__ wrow,
                             const float* __restrict__ bd, float* __restrict__ cbuf,
                             float* __restrict__ out, int n, int phase) {
    int i = blockIdx.x * blockDim.x + threadIdx.x;
    if (i >= n) return;
    const float* p = h + (size_t)i * 64;
    float s = 0.f, dsum = 0.f;
    #pragma unroll 8
    for (int f = 0; f < 64; ++f) {
        float a = p[f];
        s += a;
        dsum += a * wrow[f];
    }
    float c;
    if (phase == 0) {
        c = 1.f / (1.f + __expf(-s * (1.f / 64.f)));
        cbuf[i] = c;
        out[i] = c * dsum + bd[0];
    } else {
        c = cbuf[i];
        out[n + i] = c * dsum + bd[0];
    }
}

// ---------------------------------------------------------------------------
extern "C" void kernel_launch(void* const* d_in, const int* in_sizes, int n_in,
                              void* d_out, int out_size, void* d_ws, size_t ws_size,
                              hipStream_t stream) {
    const float* x  = (const float*)d_in[0];
    const float* cf = (const float*)d_in[1];
    const int*   ei = (const int*)d_in[2];
    const float* W1 = (const float*)d_in[3];
    const float* b1 = (const float*)d_in[4];
    const float* W2 = (const float*)d_in[5];
    const float* b2 = (const float*)d_in[6];
    const float* Wd = (const float*)d_in[7];
    const float* bd = (const float*)d_in[8];

    const int N = in_sizes[0] / 128;   // 100000
    const int E = in_sizes[2] / 2;     // 1600000
    float* out = (float*)d_out;

    // workspace layout (floats): dinv[N] | c[N] | wrow[64] | tmp[64N] | agg[64N] | hbuf[64N]
    float* ws   = (float*)d_ws;
    float* dinv = ws;
    float* cbuf = dinv + N;
    float* wrow = cbuf + N;
    float* tmp  = wrow + 64;
    float* agg  = tmp + (size_t)64 * N;
    float* hbuf = agg + (size_t)64 * N;

    const int* src = ei;
    const int* dst = ei + E;

    // --- graph normalization ---
    hipMemsetAsync(dinv, 0, (size_t)N * sizeof(float), stream);
    deg_kernel<<<(E + 255) / 256, 256, 0, stream>>>(dst, dinv, E);
    dinv_kernel<<<(N + 255) / 256, 256, 0, stream>>>(dinv, N);
    wrow_kernel<<<1, 64, 0, stream>>>(Wd, wrow);

    const long long featTot = (long long)N * 64;
    const int featBlocks    = (int)((featTot + 255) / 256);
    const int edgeBlocks    = (int)(((long long)E * 16 + 255) / 256);
    const int gemmBlocks    = (N + 15) / 16;

    for (int which = 0; which < 2; ++which) {
        const float* feat = which ? cf : x;

        // layer 1: tmp = feat @ W1 ; agg = scatter(tmp) + self + b1, ReLU
        gemm_bf16_wmma<128><<<gemmBlocks, 128, 0, stream>>>(feat, W1, tmp, N);
        hipMemsetAsync(agg, 0, (size_t)64 * N * sizeof(float), stream);
        scatter_kernel<<<edgeBlocks, 256, 0, stream>>>(tmp, src, dst, dinv, agg, E);
        postproc_kernel<<<featBlocks, 256, 0, stream>>>(agg, tmp, dinv, b1, featTot, 1);

        // layer 2: tmp = agg @ W2 ; hbuf = scatter(tmp) + self + b2
        gemm_bf16_wmma<64><<<gemmBlocks, 128, 0, stream>>>(agg, W2, tmp, N);
        hipMemsetAsync(hbuf, 0, (size_t)64 * N * sizeof(float), stream);
        scatter_kernel<<<edgeBlocks, 256, 0, stream>>>(tmp, src, dst, dinv, hbuf, E);
        postproc_kernel<<<featBlocks, 256, 0, stream>>>(hbuf, tmp, dinv, b2, featTot, 0);

        // readout
        score_kernel<<<(N + 255) / 256, 256, 0, stream>>>(hbuf, wrow, bd, cbuf, out, N, which);
    }
}
